// CausalWanSelfAttention_64433099374816
// MI455X (gfx1250) — matrix operand order repaired and checked by reading, not compiled, not run on hardware
//
#include <hip/hip_runtime.h>

#define DEVINL __device__ __forceinline__

typedef __attribute__((ext_vector_type(16))) __bf16       v16bf;
typedef __attribute__((ext_vector_type(8)))  float        v8f;
typedef __attribute__((ext_vector_type(4)))  unsigned int u32x4;

// ---- problem constants (match setup_inputs) ----
constexpr int DIM = 1536;
constexpr int NH  = 12;
constexpr int HD  = 128;
constexpr int S   = 1560;
constexpr int SINK        = 1560;
constexpr int LOCAL_START = 3120;
constexpr int LOCAL_END   = 10920;
constexpr int WRITE_START = 9360;
constexpr int KV_LEN = SINK + (LOCAL_END - LOCAL_START);   // 9360 attended keys
constexpr int KVP    = (KV_LEN + 31) & ~31;                // 9376, multiple of 32 (WMMA K)
constexpr int QT     = (S + 15) / 16;                      // 98 query tiles of 16
constexpr int NC     = DIM / 32;                           // 48 K-chunks per GEMM (even)

DEVINL unsigned short f2bfu(float f) {            // f32 -> bf16 bits, round-nearest-even
  union { float f; unsigned u; } x; x.f = f;
  unsigned u = x.u;
  unsigned r = u + 0x7fffu + ((u >> 16) & 1u);
  return (unsigned short)(r >> 16);
}

DEVINL v8f zero8() { v8f z;
#pragma unroll
  for (int i = 0; i < 8; ++i) z[i] = 0.f;
  return z;
}

DEVINL v8f wmma_bf16(v16bf a, v16bf b, v8f c) {
  // D = A(16x32 bf16) * B(32x16 bf16) + C(16x16 f32)
  return __builtin_amdgcn_wmma_f32_16x16x32_bf16(false, a, false, b, (short)0, c, false, false);
}

union Frag { v16bf v; unsigned int u[8]; u32x4 q[2]; };

// A-fragment 16x32 from row-major [row][k] bf16 (ld in elements). Contiguous k order (B uses the same):
// lane m=L%16, half=L>>4; VGPR v: K=16*half+2v+{0,1}
// The 4 dword loads per half are contiguous -> compiler merges to global_load_b128.
template <bool FULL>
DEVINL v16bf load_a_frag(const unsigned short* p, long ld, int lane, int rows_valid) {
  int m = lane & 15, half = lane >> 4;
  Frag f;
  if (FULL || m < rows_valid) {
    const unsigned short* r = p + (long)m * ld + half * 16;
#pragma unroll
    for (int v0 = 0; v0 < 4; ++v0) f.u[v0]     = *(const unsigned int*)(r + 2 * v0);
#pragma unroll
    for (int v0 = 0; v0 < 4; ++v0) f.u[4 + v0] = *(const unsigned int*)(r + 8 + 2 * v0);
  } else {
#pragma unroll
    for (int i = 0; i < 8; ++i) f.u[i] = 0u;
  }
  return f.v;
}

// B-fragment 32x16 where memory is [n][k] (each B-column contiguous along k).
// lane n=L%16 loads 16 contiguous bf16 at k-offset 16*(L>>4): two 16-byte loads.
DEVINL v16bf load_bT_frag(const unsigned short* p, long ld, int lane) {
  int n = lane & 15, koff = (lane >> 4) * 16;
  const unsigned short* r = p + (long)n * ld + koff;
  Frag f;
  f.q[0] = *(const u32x4*)(r);
  f.q[1] = *(const u32x4*)(r + 8);
  return f.v;
}

// ---------------------------------------------------------------- conversion
__global__ void k_cvt_bf16(const float* __restrict__ src, unsigned short* __restrict__ dst, int n) {
  int i = blockIdx.x * blockDim.x + threadIdx.x;
  if (i < n) dst[i] = f2bfu(src[i]);
}

// --------------------------------------------- GEMM: Y[s][o] = sum_d A[s][d]*W[o][d] + b[o]
// One wave computes a 16x128 output tile. Software-pipelined (ping-pong) over K-chunks of
// 32: chunk c+1's 18 b128 loads are issued before chunk c's 8-WMMA chain executes, so the
// loadcnt waits are covered by a full WMMA chain instead of stalling in lockstep.
DEVINL void loadB8(v16bf b[8], const unsigned short* __restrict__ W,
                   int col0, int k0, int lane) {
#pragma unroll
  for (int t = 0; t < 8; ++t)
    b[t] = load_bT_frag(W + (long)(col0 + 16 * t) * DIM + k0, DIM, lane);
}

template <bool FULL>
DEVINL void gemm_body(const unsigned short* __restrict__ A,
                      const unsigned short* __restrict__ W,
                      const float* __restrict__ bias,
                      float* __restrict__ Y,
                      int row0, int col0, int lane) {
  const unsigned short* Arow = A + (long)row0 * DIM;
  v8f acc[8];
#pragma unroll
  for (int t = 0; t < 8; ++t) acc[t] = zero8();

  v16bf aA, aB, bA[8], bB[8];
  aA = load_a_frag<FULL>(Arow, DIM, lane, S - row0);
  loadB8(bA, W, col0, 0, lane);

#pragma unroll 1
  for (int c = 0; c + 2 < NC; c += 2) {
    // issue chunk c+1 loads, then compute chunk c
    aB = load_a_frag<FULL>(Arow + (c + 1) * 32, DIM, lane, S - row0);
    loadB8(bB, W, col0, (c + 1) * 32, lane);
#pragma unroll
    for (int t = 0; t < 8; ++t) acc[t] = wmma_bf16(aA, bA[t], acc[t]);
    // issue chunk c+2 loads, then compute chunk c+1
    aA = load_a_frag<FULL>(Arow + (c + 2) * 32, DIM, lane, S - row0);
    loadB8(bA, W, col0, (c + 2) * 32, lane);
#pragma unroll
    for (int t = 0; t < 8; ++t) acc[t] = wmma_bf16(aB, bB[t], acc[t]);
  }
  // epilogue: aA/bA hold chunk NC-2; load and compute the final pair
  aB = load_a_frag<FULL>(Arow + (NC - 1) * 32, DIM, lane, S - row0);
  loadB8(bB, W, col0, (NC - 1) * 32, lane);
#pragma unroll
  for (int t = 0; t < 8; ++t) acc[t] = wmma_bf16(aA, bA[t], acc[t]);
#pragma unroll
  for (int t = 0; t < 8; ++t) acc[t] = wmma_bf16(aB, bB[t], acc[t]);

  int n = lane & 15, half = lane >> 4;
#pragma unroll
  for (int t = 0; t < 8; ++t) {
    int col = col0 + 16 * t + n;
    float bv = bias[col];
#pragma unroll
    for (int r = 0; r < 8; ++r) {
      int row = row0 + r + 8 * half;
      if (FULL || row < S) Y[(long)row * DIM + col] = acc[t][r] + bv;
    }
  }
}

__global__ __launch_bounds__(32) void k_gemm_xwT(const unsigned short* __restrict__ A,
                                                 const unsigned short* __restrict__ W,
                                                 const float* __restrict__ bias,
                                                 float* __restrict__ Y) {
  int lane = threadIdx.x;
  int row0 = blockIdx.x * 16;
  int col0 = blockIdx.y * 128;
  if (row0 + 16 <= S) gemm_body<true >(A, W, bias, Y, row0, col0, lane);
  else                gemm_body<false>(A, W, bias, Y, row0, col0, lane);
}

// ------------------- build K/V attention windows from the original cache (bf16, padded)
// kb[h][kv][d]; vt[h][d][kv] (transposed so PV B-operand is contiguous along K).
__global__ void k_build_kv(const float* __restrict__ ck, const float* __restrict__ cv,
                           unsigned short* __restrict__ kb, unsigned short* __restrict__ vt) {
  long idx = (long)blockIdx.x * blockDim.x + threadIdx.x;
  const long total = (long)NH * KVP * HD;
  if (idx >= total) return;
  int d = (int)(idx & (HD - 1));
  long t = idx >> 7;
  int r = (int)(t % KVP);
  int h = (int)(t / KVP);
  float vk = 0.f, vv = 0.f;
  if (r < KV_LEN) {
    int p = (r < SINK) ? r : (LOCAL_START + (r - SINK));       // cache position
    long src = ((long)p * NH + h) * HD + d;                    // cache layout [pos][h][d]
    vk = ck[src]; vv = cv[src];
  }
  kb[((long)h * KVP + r) * HD + d] = f2bfu(vk);
  vt[((long)h * HD + d) * KVP + r] = f2bfu(vv);
}

// -------- fused RMSNorm + RoPE epilogue; mode 0: q -> qb[h][s][d]
//          mode 1: k -> kb at spliced row; mode 2: v (no norm/rope) -> vt transposed
__global__ __launch_bounds__(256) void k_norm_rope(const float* __restrict__ y,
                                                   const float* __restrict__ nw,
                                                   const float* __restrict__ rc,
                                                   const float* __restrict__ rs,
                                                   const int* __restrict__ ropeLenP,
                                                   unsigned short* __restrict__ dst, int mode) {
  __shared__ float red[256];
  int s = blockIdx.x, tid = threadIdx.x;
  const float* row = y + (long)s * DIM;
  float rinv = 1.0f;
  if (mode < 2) {
    float ss = 0.f;
    for (int i = tid; i < DIM; i += 256) { float v = row[i]; ss += v * v; }
    red[tid] = ss; __syncthreads();
    for (int o = 128; o > 0; o >>= 1) { if (tid < o) red[tid] += red[tid + o]; __syncthreads(); }
    rinv = rsqrtf(red[0] / (float)DIM + 1e-6f);
  }
  int ropeLen = ropeLenP[0];
#pragma unroll
  for (int pp = 0; pp < 3; ++pp) {
    int pj = tid + pp * 256;              // pair index 0..767
    int h = pj >> 6, j = pj & 63;
    int d0 = 2 * j, col = h * HD + d0;
    float v0 = row[col], v1 = row[col + 1];
    if (mode < 2) {
      v0 *= rinv * nw[col]; v1 *= rinv * nw[col + 1];
      if (s < ropeLen) {
        float c = rc[s * 64 + j], sn = rs[s * 64 + j];
        float o0 = v0 * c - v1 * sn, o1 = v0 * sn + v1 * c;
        v0 = o0; v1 = o1;
      }
    }
    if (mode == 0) {
      long o = ((long)h * S + s) * HD + d0;
      dst[o] = f2bfu(v0); dst[o + 1] = f2bfu(v1);
    } else {
      int p  = WRITE_START + s;                                   // cache write position
      int kr = (p < SINK) ? p : (SINK + (p - LOCAL_START));       // row in attended window
      if (mode == 1) {
        long o = ((long)h * KVP + kr) * HD + d0;
        dst[o] = f2bfu(v0); dst[o + 1] = f2bfu(v1);
      } else {
        dst[((long)h * HD + d0) * KVP + kr]     = f2bfu(v0);
        dst[((long)h * HD + d0 + 1) * KVP + kr] = f2bfu(v1);
      }
    }
  }
}

// --------------------------------------- flash attention, one wave per (16-q-tile, head)
__global__ __launch_bounds__(32) void k_flash_attn(const unsigned short* __restrict__ qb,
                                                   const unsigned short* __restrict__ kb,
                                                   const unsigned short* __restrict__ vt,
                                                   unsigned short* __restrict__ ab) {
  __shared__ unsigned short plds[16 * 32];     // P-tile staging for D->A transpose
  int lane = threadIdx.x;
  int qt = blockIdx.x, h = blockIdx.y;
  int q0 = qt * 16;
  int n = lane & 15, half = lane >> 4;

  v16bf aq[4];
#pragma unroll
  for (int da = 0; da < 4; ++da)
    aq[da] = load_a_frag<false>(qb + ((long)h * S + q0) * HD + da * 32, HD, lane, S - q0);

  v8f acc[8];
#pragma unroll
  for (int t = 0; t < 8; ++t) acc[t] = zero8();
  float rmax[8], rsum[8];
#pragma unroll
  for (int r = 0; r < 8; ++r) { rmax[r] = -3.0e38f; rsum[r] = 0.f; }

  const float scale = 0.088388347648318447f;   // 1/sqrt(128)

  for (int kv0 = 0; kv0 < KVP; kv0 += 32) {
    const unsigned short* kbase = kb + ((long)h * KVP + kv0) * HD;
    const unsigned short* vbase = vt + ((long)h * HD) * KVP + kv0;
    __builtin_prefetch(kbase + (long)32 * HD, 0, 1);   // global_prefetch next key chunk

    // preload all 8 K fragments (one load clause), then the 8-WMMA QK^T chain
    v16bf kf[8];
#pragma unroll
    for (int da = 0; da < 4; ++da) {
      kf[da]     = load_bT_frag(kbase + da * 32, HD, lane);
      kf[4 + da] = load_bT_frag(kbase + (long)16 * HD + da * 32, HD, lane);
    }
    v8f s0 = zero8(), s1 = zero8();
#pragma unroll
    for (int da = 0; da < 4; ++da) {
      s0 = wmma_bf16(aq[da], kf[da],     s0);
      s1 = wmma_bf16(aq[da], kf[4 + da], s1);
    }

    // issue the V fragment loads now so they overlap the softmax VALU work below
    v16bf vf[8];
#pragma unroll
    for (int t = 0; t < 8; ++t)
      vf[t] = load_bT_frag(vbase + (long)(t * 16) * KVP, KVP, lane);

    int c0 = kv0 + n, c1 = c0 + 16;
    float f0[8], f1[8];
#pragma unroll
    for (int r = 0; r < 8; ++r) {
      f0[r] = (c0 < KV_LEN) ? s0[r] * scale : -1.0e30f;
      f1[r] = (c1 < KV_LEN) ? s1[r] * scale : -1.0e30f;
    }
    // online softmax; row m=r+8*half lives across the 16 lanes of this half
#pragma unroll
    for (int r = 0; r < 8; ++r) {
      float m = fmaxf(f0[r], f1[r]);
#pragma unroll
      for (int o = 1; o < 16; o <<= 1) m = fmaxf(m, __shfl_xor(m, o, 32));
      float mn  = fmaxf(rmax[r], m);
      float fac = __expf(rmax[r] - mn);
      float p0  = __expf(f0[r] - mn);
      float p1  = __expf(f1[r] - mn);
      float ls  = p0 + p1;
#pragma unroll
      for (int o = 1; o < 16; o <<= 1) ls += __shfl_xor(ls, o, 32);
      rsum[r] = rsum[r] * fac + ls;
      rmax[r] = mn;
#pragma unroll
      for (int t = 0; t < 8; ++t) acc[t][r] = acc[t][r] * fac;
      plds[(r + 8 * half) * 32 + n]      = f2bfu(p0);
      plds[(r + 8 * half) * 32 + 16 + n] = f2bfu(p1);
    }
    __syncthreads();
    v16bf ap = load_a_frag<true>(plds, 32, lane, 16);   // P in A-fragment layout
    __syncthreads();
#pragma unroll
    for (int t = 0; t < 8; ++t)                         // PV: 8 WMMA over d=128
      acc[t] = wmma_bf16(ap, vf[t], acc[t]);
  }
#pragma unroll
  for (int t = 0; t < 8; ++t) {
#pragma unroll
    for (int r = 0; r < 8; ++r) {
      int sg = q0 + r + 8 * half;
      if (sg < S) {
        int d = t * 16 + n;
        ab[(long)sg * DIM + h * HD + d] = f2bfu(acc[t][r] / rsum[r]);
      }
    }
  }
}

extern "C" void kernel_launch(void* const* d_in, const int* in_sizes, int n_in,
                              void* d_out, int out_size, void* d_ws, size_t ws_size,
                              hipStream_t stream) {
  (void)in_sizes; (void)n_in; (void)out_size; (void)ws_size;
  const float* x  = (const float*)d_in[0];
  const float* Wq = (const float*)d_in[1];
  const float* bq = (const float*)d_in[2];
  const float* Wk = (const float*)d_in[3];
  const float* bk = (const float*)d_in[4];
  const float* Wv = (const float*)d_in[5];
  const float* bv = (const float*)d_in[6];
  const float* Wo = (const float*)d_in[7];
  const float* bo = (const float*)d_in[8];
  const float* nq = (const float*)d_in[9];
  const float* nk = (const float*)d_in[10];
  const float* rc = (const float*)d_in[11];
  const float* rs = (const float*)d_in[12];
  const float* ck = (const float*)d_in[13];
  const float* cv = (const float*)d_in[14];
  const int* ropeLenP = (const int*)d_in[15];

  char* ws = (char*)d_ws;
  size_t off = 0;
  auto take = [&](size_t bytes) -> char* {
    char* p = ws + off;
    off = (off + bytes + 255) & ~(size_t)255;
    return p;
  };
  unsigned short* xb  = (unsigned short*)take((size_t)S * DIM * 2);
  unsigned short* wqb = (unsigned short*)take((size_t)DIM * DIM * 2);
  unsigned short* wkb = (unsigned short*)take((size_t)DIM * DIM * 2);
  unsigned short* wvb = (unsigned short*)take((size_t)DIM * DIM * 2);
  unsigned short* wob = (unsigned short*)take((size_t)DIM * DIM * 2);
  float* yq = (float*)take((size_t)S * DIM * 4);
  float* yk = (float*)take((size_t)S * DIM * 4);
  float* yv = (float*)take((size_t)S * DIM * 4);
  unsigned short* qb = (unsigned short*)take((size_t)NH * S * HD * 2);
  unsigned short* kb = (unsigned short*)take((size_t)NH * KVP * HD * 2);
  unsigned short* vt = (unsigned short*)take((size_t)NH * HD * KVP * 2);
  unsigned short* ab = (unsigned short*)take((size_t)S * DIM * 2);

  const int nX = S * DIM, nW = DIM * DIM;
  k_cvt_bf16<<<(nX + 255) / 256, 256, 0, stream>>>(x,  xb,  nX);
  k_cvt_bf16<<<(nW + 255) / 256, 256, 0, stream>>>(Wq, wqb, nW);
  k_cvt_bf16<<<(nW + 255) / 256, 256, 0, stream>>>(Wk, wkb, nW);
  k_cvt_bf16<<<(nW + 255) / 256, 256, 0, stream>>>(Wv, wvb, nW);
  k_cvt_bf16<<<(nW + 255) / 256, 256, 0, stream>>>(Wo, wob, nW);

  dim3 gGemm(QT, DIM / 128);
  k_gemm_xwT<<<gGemm, 32, 0, stream>>>(xb, wqb, bq, yq);
  k_gemm_xwT<<<gGemm, 32, 0, stream>>>(xb, wkb, bk, yk);
  k_gemm_xwT<<<gGemm, 32, 0, stream>>>(xb, wvb, bv, yv);

  long nKV = (long)NH * KVP * HD;
  k_build_kv<<<(int)((nKV + 255) / 256), 256, 0, stream>>>(ck, cv, kb, vt);

  k_norm_rope<<<S, 256, 0, stream>>>(yq, nq, rc, rs, ropeLenP, qb, 0);
  k_norm_rope<<<S, 256, 0, stream>>>(yk, nk, rc, rs, ropeLenP, kb, 1);
  k_norm_rope<<<S, 256, 0, stream>>>(yv, nq, rc, rs, ropeLenP, vt, 2);

  k_flash_attn<<<dim3(QT, NH), 32, 0, stream>>>(qb, kb, vt, ab);

  k_gemm_xwT<<<gGemm, 32, 0, stream>>>(ab, wob, bo, (float*)d_out);
}
